// TransformerBlock_13288628814342
// MI455X (gfx1250) — compile-verified, matrix operand order checked by reading, not run on hardware
//
#include <hip/hip_runtime.h>
#include <hip/hip_bf16.h>
#include <math.h>

typedef unsigned short u16;
typedef __attribute__((ext_vector_type(16))) __bf16 v16bf;
typedef __attribute__((ext_vector_type(8)))  float  v8f;

#define DIMC 1024
#define HEADS 16
#define HD 64
#define DFF 4096

union FragU { uint4 q[2]; v16bf v; };

__device__ __forceinline__ u16 f2bf(float f) {
  unsigned u = __float_as_uint(f);
  u += 0x7FFFu + ((u >> 16) & 1u);   // round-to-nearest-even
  return (u16)(u >> 16);
}

// A/B-style WMMA fragment: lane holds row (lane&15); elements 0-7 at
// K = (lane>>4)*8 .. +7, elements 8-15 at K = 16 + (lane>>4)*8 .. +7.
__device__ __forceinline__ v16bf load_frag(const u16* base, int half) {
  FragU f;
  f.q[0] = *(const uint4*)(base + half * 8);
  f.q[1] = *(const uint4*)(base + 16 + half * 8);
  return f.v;
}

__device__ __forceinline__ v8f wmma_bf16(v16bf a, v16bf b, v8f c) {
  return __builtin_amdgcn_wmma_f32_16x16x32_bf16(false, a, false, b,
                                                 (short)0, c, false, false);
}

// ------------------------- gfx1250 async global->LDS (ASYNCcnt) support ----
#if defined(__has_builtin)
#if __has_builtin(__builtin_amdgcn_global_load_async_to_lds_b128)
#define HAVE_ASYNC_LDS 1
#endif
#endif

#ifdef HAVE_ASYNC_LDS
typedef int v4i_vs __attribute__((vector_size(16)));
typedef __attribute__((address_space(1))) v4i_vs* gptr_b128;
typedef __attribute__((address_space(3))) v4i_vs* lptr_b128;

__device__ __forceinline__ void async_cp16(const u16* g, u16* l) {
  __builtin_amdgcn_global_load_async_to_lds_b128((gptr_b128)g, (lptr_b128)l,
                                                 0, 0);
}
__device__ __forceinline__ void wait_async0() {
#if __has_builtin(__builtin_amdgcn_s_wait_asynccnt)
  __builtin_amdgcn_s_wait_asynccnt(0);
#else
  asm volatile("s_wait_asynccnt 0x0" ::: "memory");
#endif
}
#endif

// ---------------------------------------------------------------- casts ----
__global__ void cast_bf16_kernel(const float* __restrict__ in,
                                 u16* __restrict__ out, long n) {
  long i = (long)blockIdx.x * blockDim.x + threadIdx.x;
  long stride = (long)gridDim.x * blockDim.x;
  for (; i < n; i += stride) out[i] = f2bf(in[i]);
}

// W is KxN row-major f32; Wt is NxK row-major bf16 (so GEMM B-frags are
// contiguous along K).
__global__ __launch_bounds__(256) void cast_transpose_kernel(
    const float* __restrict__ W, u16* __restrict__ Wt, int K, int N) {
  __shared__ float tile[32][33];
  const int tx = threadIdx.x & 31, ty = threadIdx.x >> 5;  // ty 0..7
  const int bx = blockIdx.x * 32;  // N
  const int by = blockIdx.y * 32;  // K
#pragma unroll
  for (int i = 0; i < 4; ++i)
    tile[ty + i * 8][tx] = W[(long)(by + ty + i * 8) * N + bx + tx];
  __syncthreads();
#pragma unroll
  for (int i = 0; i < 4; ++i)
    Wt[(long)(bx + ty + i * 8) * K + by + tx] = f2bf(tile[tx][ty + i * 8]);
}

// ----------------------------------------------------------------- GEMM ----
// C(MxN) = epi(A(MxK,bf16) @ Bt(NxK,bf16)^T + bias)
// Double-buffered: async global->LDS copies for tile i+1 overlap WMMAs on
// tile i; rotation = s_wait_asynccnt 0 + workgroup barrier.
#define BM 128
#define BN 128
#define BK 32
#define LDT (BK + 8)

enum { EPI_F32 = 0, EPI_BF16 = 1, EPI_GELU_BF16 = 2 };

template <int EPI>
__global__ __launch_bounds__(256) void gemm_bf16_kernel(
    const u16* __restrict__ A, const u16* __restrict__ Bt,
    const float* __restrict__ bias, float* __restrict__ Cf,
    u16* __restrict__ Cb, int M, int N, int K) {
  __shared__ u16 sA[2][BM * LDT];
  __shared__ u16 sB[2][BN * LDT];
  const int tid = threadIdx.x;
  const int lane = tid & 31, wave = tid >> 5;
  const int half = lane >> 4, l16 = lane & 15;
  const int wm = wave & 1, wn = wave >> 1;  // 2x4 wave grid
  const long bm = (long)blockIdx.y * BM, bn = (long)blockIdx.x * BN;

  // per-thread chunk mapping: 512 16B chunks per tile, 2 per thread
  const int row0 = tid >> 2, c80 = (tid & 3) * 8;
  const int row1 = (tid + 256) >> 2, c81 = ((tid + 256) & 3) * 8;

  auto issue = [&](int buf, int k0) {
#ifdef HAVE_ASYNC_LDS
    async_cp16(&A[(bm + row0) * (long)K + k0 + c80], &sA[buf][row0 * LDT + c80]);
    async_cp16(&Bt[(bn + row0) * (long)K + k0 + c80], &sB[buf][row0 * LDT + c80]);
    async_cp16(&A[(bm + row1) * (long)K + k0 + c81], &sA[buf][row1 * LDT + c81]);
    async_cp16(&Bt[(bn + row1) * (long)K + k0 + c81], &sB[buf][row1 * LDT + c81]);
#else
    *(uint4*)&sA[buf][row0 * LDT + c80] =
        *(const uint4*)&A[(bm + row0) * (long)K + k0 + c80];
    *(uint4*)&sB[buf][row0 * LDT + c80] =
        *(const uint4*)&Bt[(bn + row0) * (long)K + k0 + c80];
    *(uint4*)&sA[buf][row1 * LDT + c81] =
        *(const uint4*)&A[(bm + row1) * (long)K + k0 + c81];
    *(uint4*)&sB[buf][row1 * LDT + c81] =
        *(const uint4*)&Bt[(bn + row1) * (long)K + k0 + c81];
#endif
  };

  const v8f vzero = {};
  v8f acc[4][2];
#pragma unroll
  for (int i = 0; i < 4; ++i)
#pragma unroll
    for (int j = 0; j < 2; ++j) acc[i][j] = vzero;

  issue(0, 0);
  int buf = 0;
  for (int k0 = 0; k0 < K; k0 += BK) {
#ifdef HAVE_ASYNC_LDS
    wait_async0();
#endif
    __syncthreads();  // tile `buf` resident & visible to all waves
    if (k0 + BK < K) issue(buf ^ 1, k0 + BK);  // overlap next tile with math

    v16bf af[4], bfr[2];
#pragma unroll
    for (int i = 0; i < 4; ++i)
      af[i] = load_frag(&sA[buf][(wm * 64 + i * 16 + l16) * LDT], half);
#pragma unroll
    for (int j = 0; j < 2; ++j)
      bfr[j] = load_frag(&sB[buf][(wn * 32 + j * 16 + l16) * LDT], half);
#pragma unroll
    for (int i = 0; i < 4; ++i)
#pragma unroll
      for (int j = 0; j < 2; ++j)
        acc[i][j] = wmma_bf16(af[i], bfr[j], acc[i][j]);

    __syncthreads();  // all reads of `buf` done before it is refilled
    buf ^= 1;
  }

  // C layout: element r of v8f -> M = r + 8*half, N = l16
#pragma unroll
  for (int i = 0; i < 4; ++i)
#pragma unroll
    for (int j = 0; j < 2; ++j) {
      long col = bn + wn * 32 + j * 16 + l16;
      float bv = bias[col];
#pragma unroll
      for (int r = 0; r < 8; ++r) {
        long row = bm + wm * 64 + i * 16 + half * 8 + r;
        float v = acc[i][j][r] + bv;
        if (EPI == EPI_F32) {
          Cf[row * N + col] = v;
        } else if (EPI == EPI_BF16) {
          Cb[row * N + col] = f2bf(v);
        } else {  // exact GELU
          float g = 0.5f * v * (1.0f + erff(v * 0.70710678118654752f));
          Cb[row * N + col] = f2bf(g);
        }
      }
    }
}

// ------------------------------------------------------- flash attention ---
// qkv: (B*T, 3*DIMC) bf16 laid out [comp][head][d]; o: (B*T, DIMC) bf16.
#define ALD 72
__global__ __launch_bounds__(128) void attn_kernel(
    const u16* __restrict__ qkv, u16* __restrict__ o, int B, int T) {
  __shared__ u16 sVt[HD * ALD];      // V^T tile: [d][key]
  __shared__ u16 sP[4 * 16 * ALD];   // per-wave P tile (C->A relayout)
  const int tid = threadIdx.x;
  const int lane = tid & 31, wave = tid >> 5;
  const int half = lane >> 4, l16 = lane & 15;
  const int bh = blockIdx.y;
  const int b = bh >> 4, h = bh & 15;
  const int qt = blockIdx.x;
  const long rs = 3 * DIMC;  // qkv row stride (tokens)
  const u16* qb = qkv + (long)b * T * rs + (long)h * HD;
  const u16* kb = qb + DIMC;
  const u16* vb = qb + 2 * DIMC;

  // Q fragments straight from global (head dim contiguous, 16B aligned)
  const int q0 = qt * 64 + wave * 16;
  v16bf aq[2];
#pragma unroll
  for (int kc = 0; kc < 2; ++kc)
    aq[kc] = load_frag(qb + (long)(q0 + l16) * rs + kc * 32, half);

  const v8f vzero = {};
  float mrow[8], lrow[8];
  v8f accO[4];
#pragma unroll
  for (int r = 0; r < 8; ++r) { mrow[r] = -1e30f; lrow[r] = 0.f; }
#pragma unroll
  for (int j = 0; j < 4; ++j) accO[j] = vzero;

  for (int kt = 0; kt < T / 64; ++kt) {
    // cooperative transpose of the 64x64 V tile into LDS
#pragma unroll
    for (int i = 0; i < 4; ++i) {
      int chunk = tid + i * 128;  // 512 chunks of 8 bf16
      int key = chunk >> 3, d8 = (chunk & 7) * 8;
      uint4 raw = *(const uint4*)(vb + (long)(kt * 64 + key) * rs + d8);
      const u16* pv = (const u16*)&raw;
#pragma unroll
      for (int e = 0; e < 8; ++e) sVt[(d8 + e) * ALD + key] = pv[e];
    }
    __syncthreads();

    // S = Q K^T ; K^T B-frag == A-style frag of K rows (contiguous d)
    v8f accS[4];
#pragma unroll
    for (int j = 0; j < 4; ++j) {
      accS[j] = vzero;
#pragma unroll
      for (int kc = 0; kc < 2; ++kc) {
        v16bf bk =
            load_frag(kb + (long)(kt * 64 + j * 16 + l16) * rs + kc * 32, half);
        accS[j] = wmma_bf16(aq[kc], bk, accS[j]);
      }
    }

    // online softmax: rows live in 16-lane groups (lane>>4 fixed)
    float nm[8];
#pragma unroll
    for (int r = 0; r < 8; ++r) nm[r] = mrow[r];
#pragma unroll
    for (int j = 0; j < 4; ++j)
#pragma unroll
      for (int r = 0; r < 8; ++r) {
        float s = accS[j][r] * 0.125f;  // 1/sqrt(64)
        accS[j][r] = s;
        nm[r] = fmaxf(nm[r], s);
      }
#pragma unroll
    for (int off = 1; off < 16; off <<= 1)
#pragma unroll
      for (int r = 0; r < 8; ++r)
        nm[r] = fmaxf(nm[r], __shfl_xor(nm[r], off, 32));

    float alpha[8], rsum[8];
#pragma unroll
    for (int r = 0; r < 8; ++r) {
      alpha[r] = __expf(mrow[r] - nm[r]);
      mrow[r] = nm[r];
      rsum[r] = 0.f;
    }
#pragma unroll
    for (int j = 0; j < 4; ++j)
#pragma unroll
      for (int r = 0; r < 8; ++r) {
        float p = __expf(accS[j][r] - mrow[r]);
        rsum[r] += p;
        sP[(wave * 16 + half * 8 + r) * ALD + j * 16 + l16] = f2bf(p);
      }
#pragma unroll
    for (int off = 1; off < 16; off <<= 1)
#pragma unroll
      for (int r = 0; r < 8; ++r) rsum[r] += __shfl_xor(rsum[r], off, 32);
#pragma unroll
    for (int r = 0; r < 8; ++r) lrow[r] = lrow[r] * alpha[r] + rsum[r];
#pragma unroll
    for (int j = 0; j < 4; ++j)
#pragma unroll
      for (int r = 0; r < 8; ++r) accO[j][r] = accO[j][r] * alpha[r];

    // O += P @ V  (same-wave LDS is in-order: no barrier needed for sP)
#pragma unroll
    for (int kc = 0; kc < 2; ++kc) {
      v16bf ap = load_frag(&sP[(wave * 16 + l16) * ALD + kc * 32], half);
#pragma unroll
      for (int j = 0; j < 4; ++j) {
        v16bf bv = load_frag(&sVt[(j * 16 + l16) * ALD + kc * 32], half);
        accO[j] = wmma_bf16(ap, bv, accO[j]);
      }
    }
    __syncthreads();  // protect sVt before next tile
  }

#pragma unroll
  for (int j = 0; j < 4; ++j)
#pragma unroll
    for (int r = 0; r < 8; ++r) {
      float v = accO[j][r] / lrow[r];
      long row = (long)b * T + qt * 64 + wave * 16 + half * 8 + r;
      o[row * DIMC + h * HD + j * 16 + l16] = f2bf(v);
    }
}

// -------------------------------------------------- residual + layernorm ---
__global__ __launch_bounds__(256) void add_ln_kernel(
    const float* __restrict__ x, const float* __restrict__ y,
    const float* __restrict__ g, const float* __restrict__ be,
    float* __restrict__ outF, u16* __restrict__ outB) {
  const int row = blockIdx.x;
  const int tid = threadIdx.x;
  const int lane = tid & 31, wave = tid >> 5;
  const float* yr = y + (long)row * DIMC;
  __shared__ float red1[8], red2[8];

  float v[4];
  float s = 0.f;
#pragma unroll
  for (int i = 0; i < 4; ++i) { v[i] = yr[tid + i * 256]; s += v[i]; }
#pragma unroll
  for (int off = 1; off < 32; off <<= 1) s += __shfl_xor(s, off, 32);
  if (lane == 0) red1[wave] = s;
  __syncthreads();
  float mu = 0.f;
#pragma unroll
  for (int w = 0; w < 8; ++w) mu += red1[w];
  mu *= (1.0f / DIMC);

  float t = 0.f;
#pragma unroll
  for (int i = 0; i < 4; ++i) { float d = v[i] - mu; t += d * d; }
#pragma unroll
  for (int off = 1; off < 32; off <<= 1) t += __shfl_xor(t, off, 32);
  if (lane == 0) red2[wave] = t;
  __syncthreads();
  float var = 0.f;
#pragma unroll
  for (int w = 0; w < 8; ++w) var += red2[w];
  var *= (1.0f / DIMC);
  float rsv = rsqrtf(var + 1e-5f);

#pragma unroll
  for (int i = 0; i < 4; ++i) {
    int c = tid + i * 256;
    float ov = x[(long)row * DIMC + c] + (v[i] - mu) * rsv * g[c] + be[c];
    if (outF) outF[(long)row * DIMC + c] = ov;
    if (outB) outB[(long)row * DIMC + c] = f2bf(ov);
  }
}

// ---------------------------------------------------------------- driver ---
extern "C" void kernel_launch(void* const* d_in, const int* in_sizes, int n_in,
                              void* d_out, int out_size, void* d_ws,
                              size_t ws_size, hipStream_t stream) {
  const float* x    = (const float*)d_in[0];
  const float* Wqkv = (const float*)d_in[1];
  const float* bqkv = (const float*)d_in[2];
  const float* Wout = (const float*)d_in[3];
  const float* bout = (const float*)d_in[4];
  const float* W1   = (const float*)d_in[5];
  const float* b1   = (const float*)d_in[6];
  const float* W2   = (const float*)d_in[7];
  const float* b2   = (const float*)d_in[8];
  const float* g1   = (const float*)d_in[9];
  const float* be1  = (const float*)d_in[10];
  const float* g2   = (const float*)d_in[11];
  const float* be2  = (const float*)d_in[12];
  (void)in_sizes; (void)n_in; (void)out_size; (void)ws_size;

  const int Bsz = 2, T = 2048;
  const long NT = (long)Bsz * T;

  char* ws = (char*)d_ws;
  size_t off = 0;
  auto alloc = [&](size_t bytes) -> void* {
    void* p = ws + off;
    off += (bytes + 255) & ~(size_t)255;
    return p;
  };
  u16*   xb    = (u16*)alloc((size_t)NT * DIMC * 2);
  u16*   WqkvT = (u16*)alloc((size_t)3 * DIMC * DIMC * 2);
  u16*   WoutT = (u16*)alloc((size_t)DIMC * DIMC * 2);
  u16*   W1T   = (u16*)alloc((size_t)DFF * DIMC * 2);
  u16*   W2T   = (u16*)alloc((size_t)DIMC * DFF * 2);
  u16*   qkvb  = (u16*)alloc((size_t)NT * 3 * DIMC * 2);
  u16*   ob    = (u16*)alloc((size_t)NT * DIMC * 2);
  float* attnF = (float*)alloc((size_t)NT * DIMC * 4);
  float* x1F   = (float*)alloc((size_t)NT * DIMC * 4);
  u16*   x1B   = (u16*)alloc((size_t)NT * DIMC * 2);
  u16*   hB    = (u16*)alloc((size_t)NT * DFF * 2);
  float* ffnF  = (float*)alloc((size_t)NT * DIMC * 4);

  // precision staging
  cast_bf16_kernel<<<1024, 256, 0, stream>>>(x, xb, NT * DIMC);
  cast_transpose_kernel<<<dim3(3 * DIMC / 32, DIMC / 32), 256, 0, stream>>>(
      Wqkv, WqkvT, DIMC, 3 * DIMC);
  cast_transpose_kernel<<<dim3(DIMC / 32, DIMC / 32), 256, 0, stream>>>(
      Wout, WoutT, DIMC, DIMC);
  cast_transpose_kernel<<<dim3(DFF / 32, DIMC / 32), 256, 0, stream>>>(
      W1, W1T, DIMC, DFF);
  cast_transpose_kernel<<<dim3(DIMC / 32, DFF / 32), 256, 0, stream>>>(
      W2, W2T, DFF, DIMC);

  // qkv = x @ Wqkv + bqkv   (bf16 out, consumed by attention)
  gemm_bf16_kernel<EPI_BF16>
      <<<dim3(3 * DIMC / BN, NT / BM), 256, 0, stream>>>(
          xb, WqkvT, bqkv, nullptr, qkvb, (int)NT, 3 * DIMC, DIMC);
  // flash attention -> concat-head bf16 output
  attn_kernel<<<dim3(T / 64, Bsz * HEADS), 128, 0, stream>>>(qkvb, ob, Bsz, T);
  // attn_out = o @ Wout + bout (f32 for LN)
  gemm_bf16_kernel<EPI_F32><<<dim3(DIMC / BN, NT / BM), 256, 0, stream>>>(
      ob, WoutT, bout, attnF, nullptr, (int)NT, DIMC, DIMC);
  // x1 = x + LN(attn_out)
  add_ln_kernel<<<(int)NT, 256, 0, stream>>>(x, attnF, g1, be1, x1F, x1B);
  // h = gelu(x1 @ W1 + b1) (bf16)
  gemm_bf16_kernel<EPI_GELU_BF16>
      <<<dim3(DFF / BN, NT / BM), 256, 0, stream>>>(x1B, W1T, b1, nullptr, hB,
                                                    (int)NT, DFF, DIMC);
  // ffn_out = h @ W2 + b2 (f32)
  gemm_bf16_kernel<EPI_F32><<<dim3(DIMC / BN, NT / BM), 256, 0, stream>>>(
      hB, W2T, b2, ffnF, nullptr, (int)NT, DIMC, DFF);
  // out = x1 + LN(ffn_out)
  add_ln_kernel<<<(int)NT, 256, 0, stream>>>(x1F, ffnF, g2, be2, (float*)d_out,
                                             nullptr);
}
